// S4D_74139725463704
// MI455X (gfx1250) — matrix operand order repaired and checked by reading, not compiled, not run on hardware
//
#include <hip/hip_runtime.h>
#include <hip/hip_bf16.h>

// S4D fused pipeline for MI455X (gfx1250), wave32 + WMMA bf16.
// Causal long-conv done as block-Toeplitz GEMM (N dim = batch = 16),
// GLU pointwise conv done as dual-accumulator bf16 WMMA GEMM.

#define L_SEQ 2048
#define B_SZ  16
#define H_DIM 512
#define N_HALF 32
#define KPAD  64
#define KROW  (KPAD + L_SEQ)   // zero-padded kernel row

typedef __attribute__((ext_vector_type(16))) __bf16 v16bf;
typedef __attribute__((ext_vector_type(8)))  __bf16 v8bf;
typedef __attribute__((ext_vector_type(8)))  float  v8f;

// ---------------------------------------------------------------------------
// Kernel 1: S4D mode prep + Vandermonde kernel materialization (bf16, padded)
// k[h,l] = 2*Re( sum_n Cm_n * exp(dtA_n * l) ),  Cm = C * (exp(dtA)-1)/A
// ---------------------------------------------------------------------------
__global__ void __launch_bounds__(256) s4d_prep_kernel(
    const float* __restrict__ log_dt, const float* __restrict__ C,
    const float* __restrict__ log_A_real, const float* __restrict__ A_imag,
    __bf16* __restrict__ kpad)
{
  __shared__ float sRe[N_HALF], sIm[N_HALF], sCr[N_HALF], sCi[N_HALF];
  const int h = blockIdx.x;
  const int t = threadIdx.x;
  if (t < N_HALF) {
    float dt  = expf(log_dt[h]);
    float Are = -expf(log_A_real[h * N_HALF + t]);
    float Aim = A_imag[h * N_HALF + t];
    float dre = Are * dt, dim = Aim * dt;
    float er  = expf(dre);
    float cs, sn; sincosf(dim, &sn, &cs);
    float lre = er * cs, lim = er * sn;          // exp(dtA)
    float nre = lre - 1.f, nim = lim;            // exp(dtA)-1
    float inv = 1.f / (Are * Are + Aim * Aim);
    float tre = (nre * Are + nim * Aim) * inv;   // (exp(dtA)-1)/A
    float tim = (nim * Are - nre * Aim) * inv;
    float Ccr = C[(h * N_HALF + t) * 2 + 0];
    float Cci = C[(h * N_HALF + t) * 2 + 1];
    sCr[t] = 2.f * (Ccr * tre - Cci * tim);      // fold the 2*Re() scale in
    sCi[t] = 2.f * (Ccr * tim + Cci * tre);
    sRe[t] = dre; sIm[t] = dim;
  }
  __syncthreads();
  __bf16* row = kpad + (size_t)h * KROW;
  if (t < KPAD) row[t] = (__bf16)0.f;            // causal zero pad
  for (int l = t; l < L_SEQ; l += 256) {
    float fl = (float)l, acc = 0.f;
    #pragma unroll 4
    for (int n = 0; n < N_HALF; ++n) {
      float mag = expf(sRe[n] * fl);
      float cs, sn; sincosf(sIm[n] * fl, &sn, &cs);
      acc += mag * (sCr[n] * cs - sCi[n] * sn);
    }
    row[KPAD + l] = (__bf16)acc;
  }
}

// ---------------------------------------------------------------------------
// Kernel 2: LayerNorm over H, one wave32 per (l,b) row; writes un bf16 in
// (H, B, L) layout (B-operand friendly for the Toeplitz WMMA, L2-resident).
// ---------------------------------------------------------------------------
__global__ void __launch_bounds__(256) s4d_ln_kernel(
    const float* __restrict__ u, const float* __restrict__ lnw,
    const float* __restrict__ lnb, __bf16* __restrict__ un_t)
{
  const int lane = threadIdx.x & 31, warp = threadIdx.x >> 5;
  const int row  = blockIdx.x * 8 + warp;        // row = l*B + b
  const float* x = u + (size_t)row * H_DIM;
  float xv[16]; float s = 0.f;
  #pragma unroll
  for (int j = 0; j < 16; ++j) { xv[j] = x[lane + 32 * j]; s += xv[j]; }
  #pragma unroll
  for (int m = 16; m; m >>= 1) s += __shfl_xor(s, m, 32);
  const float mean = s * (1.f / H_DIM);
  float v = 0.f;
  #pragma unroll
  for (int j = 0; j < 16; ++j) { float d = xv[j] - mean; v += d * d; }
  #pragma unroll
  for (int m = 16; m; m >>= 1) v += __shfl_xor(v, m, 32);
  const float inv = rsqrtf(v * (1.f / H_DIM) + 1e-5f);
  const int l = row >> 4, b = row & 15;
  #pragma unroll
  for (int j = 0; j < 16; ++j) {
    int h = lane + 32 * j;
    float y = (xv[j] - mean) * inv * lnw[h] + lnb[h];
    un_t[(size_t)h * (B_SZ * L_SEQ) + (size_t)b * L_SEQ + l] = (__bf16)y;
  }
}

// ---------------------------------------------------------------------------
// Kernel 3: causal conv as block-Toeplitz WMMA GEMM, one block per h.
// Output tile: M=16 seq positions x N=16 batches; K=32 source positions.
// A[m,K] = k[p+m - (k0+K)]  (zero pad handles causality + ragged edge).
// Loop bounds are readfirstlane'd into SGPRs: scalar loop control, EXEC is
// provably all-ones at the WMMAs (ISA requirement), no exec-masked loops.
// Epilogue: + un*D[h], exact GELU, store y_gelu bf16 as (B*L, H).
// ---------------------------------------------------------------------------
__global__ void __launch_bounds__(256) s4d_conv_kernel(
    const __bf16* __restrict__ kpad, const __bf16* __restrict__ un_t,
    const float* __restrict__ Dp, __bf16* __restrict__ yg)
{
  __shared__ __bf16 kl[KROW];
  const int h = blockIdx.x;
  for (int i = threadIdx.x; i < KROW; i += 256)
    kl[i] = kpad[(size_t)h * KROW + i];
  __syncthreads();
  const float Dh = Dp[h];
  const int lane = threadIdx.x & 31;
  const int warp = __builtin_amdgcn_readfirstlane(threadIdx.x >> 5); // SGPR
  const int m = lane & 15, hi = lane >> 4;
  const int kbase = hi * 8;        // ISA 16-bit A layout: K group per half-wave
  const int koff  = hi * 16;       // B layout: K half per half-wave
  const int b = lane & 15;         // batch = N column
  const __bf16* ub = un_t + (size_t)h * (B_SZ * L_SEQ) + (size_t)b * L_SEQ;

  for (int tile = warp; tile < L_SEQ / 16; tile += 8) {
    const int p = tile * 16;       // uniform (SGPR-derived)
    v8f acc = {};
    #pragma unroll 2
    for (int k0 = 0; k0 <= p; k0 += 32) {
      v16bf Bv = *(const v16bf*)&ub[k0 + koff];          // 32B aligned
      v16bf Av;
      const int base = KPAD + p + m - k0;
      #pragma unroll
      for (int e = 0; e < 8; ++e) Av[e]     = kl[base - (koff + e)];
      #pragma unroll
      for (int e = 0; e < 8; ++e) Av[8 + e] = kl[base - (koff + 8 + e)];
      acc = __builtin_amdgcn_wmma_f32_16x16x32_bf16(
          false, Av, false, Bv, (short)0, acc, false, false);
    }
    // Epilogue: one vector load for the 8 contiguous un values per lane.
    v8bf uv = *(const v8bf*)&ub[p + 8 * hi];             // 16B aligned
    #pragma unroll
    for (int r = 0; r < 8; ++r) {
      const int l = p + r + 8 * hi;
      float y = acc[r] + (float)uv[r] * Dh;              // D skip
      y = 0.5f * y * (1.f + erff(y * 0.70710678118654752f));  // exact GELU
      yg[((size_t)b * L_SEQ + l) * H_DIM + h] = (__bf16)y;
    }
  }
}

// ---------------------------------------------------------------------------
// Kernel 4: W (2H x H, f32) -> bf16
// ---------------------------------------------------------------------------
__global__ void __launch_bounds__(256) s4d_wcvt_kernel(
    const float* __restrict__ W, __bf16* __restrict__ Wb)
{
  int i = blockIdx.x * 256 + threadIdx.x;
  Wb[i] = (__bf16)W[i];
}

// ---------------------------------------------------------------------------
// Kernel 5: GLU GEMM.  v = y_gelu @ W^T + b;  out = u + a * sigmoid(g).
// Each wave: one 16x16 M/N tile with TWO accumulators (a-half and g-half of
// the 2H output) sharing the A operand across the K=512 loop.
// ---------------------------------------------------------------------------
__global__ void __launch_bounds__(256) s4d_glu_kernel(
    const __bf16* __restrict__ yg, const __bf16* __restrict__ Wb,
    const float* __restrict__ bconv, const float* __restrict__ u,
    float* __restrict__ out)
{
  const int lane = threadIdx.x & 31;
  const int warp = __builtin_amdgcn_readfirstlane(threadIdx.x >> 5);
  const int wid = blockIdx.x * 8 + warp;      // 65536 wave-tiles (SGPR)
  const int nt = wid & 31, mt = wid >> 5;     // 32 N-tiles x 2048 M-tiles
  const int r0 = mt * 16, o0 = nt * 16;
  const int m = lane & 15, hi = lane >> 4;
  const int kbase = hi * 8, koff = hi * 16, n = lane & 15;
  const __bf16* arow   = yg + (size_t)(r0 + m) * H_DIM;
  const __bf16* browA  = Wb + (size_t)(o0 + n) * H_DIM;
  const __bf16* browG  = Wb + (size_t)(o0 + n + H_DIM) * H_DIM;
  v8f accA = {}, accG = {};
  #pragma unroll 4
  for (int h0 = 0; h0 < H_DIM; h0 += 32) {
    __builtin_prefetch(&arow[h0 + 64], 0, 0);            // global_prefetch_b8
    union { v16bf v; v8bf p[2]; } Au;
    Au.p[0] = *(const v8bf*)&arow[h0 + koff];            // 16B aligned
    Au.p[1] = *(const v8bf*)&arow[h0 + koff + 8];
    v16bf Ba = *(const v16bf*)&browA[h0 + koff];         // 32B aligned
    v16bf Bg = *(const v16bf*)&browG[h0 + koff];
    accA = __builtin_amdgcn_wmma_f32_16x16x32_bf16(
        false, Au.v, false, Ba, (short)0, accA, false, false);
    accG = __builtin_amdgcn_wmma_f32_16x16x32_bf16(
        false, Au.v, false, Bg, (short)0, accG, false, false);
  }
  const float ba = bconv[o0 + n], bg = bconv[H_DIM + o0 + n];
  #pragma unroll
  for (int r = 0; r < 8; ++r) {
    const int row = r0 + r + 8 * hi;          // = b*L + l
    const int bI = row >> 11, l = row & (L_SEQ - 1);
    const size_t idx = ((size_t)l * B_SZ + bI) * H_DIM + (o0 + n);
    float a = accA[r] + ba;
    float g = accG[r] + bg;
    out[idx] = u[idx] + a * (1.f / (1.f + expf(-g)));    // residual + GLU
  }
}

// ---------------------------------------------------------------------------
// Launcher. Workspace layout (all L2-resident on MI455X, 192MB L2):
//   kpad  bf16  H x 2112                 = 2,162,688 B
//   un_t  bf16  (H,B,L)                  = 33,554,432 B
//   yg    bf16  (B*L, H)                 = 33,554,432 B
//   Wb    bf16  (2H, H)                  =  1,048,576 B
// ---------------------------------------------------------------------------
extern "C" void kernel_launch(void* const* d_in, const int* in_sizes, int n_in,
                              void* d_out, int out_size, void* d_ws, size_t ws_size,
                              hipStream_t stream) {
  (void)in_sizes; (void)n_in; (void)out_size; (void)ws_size;
  const float* u      = (const float*)d_in[0];
  const float* log_dt = (const float*)d_in[1];
  const float* C      = (const float*)d_in[2];
  const float* logAr  = (const float*)d_in[3];
  const float* Aim    = (const float*)d_in[4];
  const float* Dp     = (const float*)d_in[5];
  const float* W      = (const float*)d_in[6];
  const float* bconv  = (const float*)d_in[7];
  const float* lnw    = (const float*)d_in[8];
  const float* lnb    = (const float*)d_in[9];

  char* ws = (char*)d_ws;
  __bf16* kpad = (__bf16*)(ws);
  __bf16* un_t = (__bf16*)(ws + 2162688);
  __bf16* yg   = (__bf16*)(ws + 2162688 + 33554432);
  __bf16* Wb   = (__bf16*)(ws + 2162688 + 33554432 + 33554432);

  s4d_prep_kernel<<<H_DIM, 256, 0, stream>>>(log_dt, C, logAr, Aim, kpad);
  s4d_ln_kernel<<<(L_SEQ * B_SZ) / 8, 256, 0, stream>>>(u, lnw, lnb, un_t);
  s4d_wcvt_kernel<<<(2 * H_DIM * H_DIM) / 256, 256, 0, stream>>>(W, Wb);
  s4d_conv_kernel<<<H_DIM, 256, 0, stream>>>(kpad, un_t, Dp, yg);
  s4d_glu_kernel<<<(2048 * 32) / 8, 256, 0, stream>>>(yg, Wb, bconv, u,
                                                      (float*)d_out);
}
